// DMR_FlotEncoder_67327907332135
// MI455X (gfx1250) — matrix operand order repaired and checked by reading, not compiled, not run on hardware
//
#include <hip/hip_runtime.h>
#include <hip/hip_bf16.h>

typedef __bf16 bf16_t;
typedef __attribute__((ext_vector_type(16))) bf16_t v16bf;
typedef __attribute__((ext_vector_type(8)))  bf16_t v8bf;
typedef __attribute__((ext_vector_type(8)))  float  v8f;

#define NPTS 4096
#define KNN  32

// ---------------------------------------------------------------------------
// Cross-lane helpers (pure VALU, no LDS):
//   fxor<mask> : lane i gets value of lane i^mask, mask in {1,2,4,8}
//                (DPP16 row_xmask, stays inside a 16-lane row)
//   fxor16     : lane i gets value of lane i^16 (v_permlanex16, identity sel)
// ---------------------------------------------------------------------------
template<int MASK>
__device__ __forceinline__ float fxor(float x) {
  static_assert(MASK == 1 || MASK == 2 || MASK == 4 || MASK == 8, "row_xmask");
  int r = __builtin_amdgcn_update_dpp(0, __builtin_bit_cast(int, x),
                                      0x160 | MASK, 0xF, 0xF, true);
  return __builtin_bit_cast(float, r);
}
__device__ __forceinline__ float fxor16(float x) {
  int xi = __builtin_bit_cast(int, x);
  int r = __builtin_amdgcn_permlanex16(xi, xi, 0x76543210, 0xfedcba98,
                                       false, false);
  return __builtin_bit_cast(float, r);
}

// ---------------------------------------------------------------------------
// Weight prep: (cout, d) f32 -> (cout, dpad) bf16, zero padded in K dim.
// ---------------------------------------------------------------------------
__global__ void pad_weight(const float* __restrict__ w, bf16_t* __restrict__ out,
                           int cout, int d, int dpad) {
  int i = blockIdx.x * blockDim.x + threadIdx.x;
  int total = cout * dpad;
  if (i >= total) return;
  int c = i / dpad, k = i - c * dpad;
  out[i] = (k < d) ? (bf16_t)w[c * d + k] : (bf16_t)0.0f;
}

// ---------------------------------------------------------------------------
// KNN: one thread per query point; candidates staged via LDS in 512-pt chunks.
// ---------------------------------------------------------------------------
__global__ __launch_bounds__(256) void knn_kernel(const float* __restrict__ pc,
                                                  int* __restrict__ idx) {
  __shared__ float sp[512 * 3];
  const int tid = threadIdx.x;
  const int pid = blockIdx.x * 256 + tid;       // all threads in block share b
  const int b = pid >> 12;
  const float qx = pc[(size_t)pid * 3 + 0];
  const float qy = pc[(size_t)pid * 3 + 1];
  const float qz = pc[(size_t)pid * 3 + 2];

  float bd[KNN];
  int   bi[KNN];
  for (int i = 0; i < KNN; ++i) { bd[i] = 3.0e38f; bi[i] = 0; }
  float wmax = 3.0e38f;
  int   wslot = 0;

  const float* pb = pc + (size_t)b * NPTS * 3;
  for (int mb = 0; mb < NPTS; mb += 512) {
    __syncthreads();
    for (int t = tid; t < 512 * 3; t += 256) sp[t] = pb[(size_t)mb * 3 + t];
    __syncthreads();
    for (int mm = 0; mm < 512; ++mm) {
      float dx = sp[mm * 3 + 0] - qx;
      float dy = sp[mm * 3 + 1] - qy;
      float dz = sp[mm * 3 + 2] - qz;
      float d = dx * dx + dy * dy + dz * dz;
      if (d < wmax) {
        bd[wslot] = d; bi[wslot] = mb + mm;
        wmax = -1.0f;
        for (int i = 0; i < KNN; ++i)
          if (bd[i] > wmax) { wmax = bd[i]; wslot = i; }
      }
    }
  }
  for (int i = 0; i < KNN; ++i) idx[(size_t)pid * KNN + i] = bi[i];
}

// ---------------------------------------------------------------------------
// One (linear -> GroupNorm(8) -> LeakyReLU(0.1)) sub-layer for one point,
// executed by a single wave32.  H(32 x DIN) @ W^T(DIN x C), WMMA bf16.
// A fragments from LDS, B fragments from global bf16 weights.
// GroupNorm stats via DPP row_xmask butterflies (no LDS traffic).
// ---------------------------------------------------------------------------
template<int DIN, int C, int MAXD, bool FINAL>
__device__ __forceinline__ void run_sublayer(
    const bf16_t* __restrict__ A, bf16_t* __restrict__ Bout,
    const bf16_t* __restrict__ W,
    const float* __restrict__ bias, const float* __restrict__ gw,
    const float* __restrict__ gb,
    float* outp, int ostride, int lane) {
  constexpr int GS = C / 8;             // group-norm group size: 4 / 8 / 16
  const int lmod = lane & 15;
  const int lhi  = lane >> 4;           // 0 or 1

  for (int tc = 0; tc < C / 16; ++tc) {
    const int col = tc * 16 + lmod;
    const float bcol = bias[col];
    const float gcol = gw[col];
    const float gbc  = gb[col];
    const bf16_t* wrow = W + (size_t)col * DIN + lhi * 16;
    float colmax = -3.0e38f;

#pragma unroll
    for (int tm = 0; tm < 2; ++tm) {
      v8f acc = {0.f, 0.f, 0.f, 0.f, 0.f, 0.f, 0.f, 0.f};
      const bf16_t* arow = A + (size_t)(tm * 16 + lmod) * MAXD + lhi * 16;
#pragma unroll
      for (int kk = 0; kk < DIN / 32; ++kk) {
        // A (16x32 bf16): lane holds row M=l%16; k = kk*32 + (l/16)*16 + {0..7}
        // and kk*32 + (l/16)*16 + 8 + {0..7}  -> two 16-byte LDS loads.
        v8bf alo = *(const v8bf*)(arow + kk * 32);
        v8bf ahi = *(const v8bf*)(arow + kk * 32 + 8);
        v16bf av;
#pragma unroll
        for (int i = 0; i < 8; ++i) { av[i] = alo[i]; av[i + 8] = ahi[i]; }
        // B (32x16 bf16): lane holds col N=l%16; k = kk*32 + (l/16)*16 + {0..15}
        // -> one 32-byte contiguous global load.
        v16bf bv = *(const v16bf*)(wrow + kk * 32);
        if (kk + 1 < DIN / 32) __builtin_prefetch(wrow + (kk + 1) * 32, 0, 1);
        acc = __builtin_amdgcn_wmma_f32_16x16x32_bf16(
            false, av, false, bv, (short)0, acc, false, false);
      }
      // bias + GroupNorm + LeakyReLU. acc[v] is (row m = tm*16 + lhi*8 + v, col).
      float tmax = -3.0e38f;
#pragma unroll
      for (int v = 0; v < 8; ++v) {
        float x = acc[v] + bcol;
        float s = x, s2 = x * x;
        // butterfly over the group's columns (lane-xor masks < 16 -> DPP)
        s += fxor<1>(s);  s2 += fxor<1>(s2);
        s += fxor<2>(s);  s2 += fxor<2>(s2);
        if constexpr (GS >= 8)  { s += fxor<4>(s);  s2 += fxor<4>(s2); }
        if constexpr (GS >= 16) { s += fxor<8>(s);  s2 += fxor<8>(s2); }
        float mean = s * (1.0f / GS);
        float var  = s2 * (1.0f / GS) - mean * mean;
        float y = (x - mean) * rsqrtf(var + 1e-5f) * gcol + gbc;
        y = (y > 0.0f) ? y : 0.1f * y;
        if constexpr (!FINAL) {
          const int m = tm * 16 + lhi * 8 + v;
          Bout[(size_t)m * MAXD + col] = (bf16_t)y;
        } else {
          tmax = fmaxf(tmax, y);
        }
      }
      if constexpr (FINAL) colmax = fmaxf(colmax, tmax);
    }
    if constexpr (FINAL) {      // max over all 32 neighbor rows, then store
      colmax = fmaxf(colmax, fxor16(colmax));
      if (lane < 16) outp[(size_t)col * ostride] = colmax;
    }
  }
}

// ---------------------------------------------------------------------------
// One EdgeConv layer: gather + epos concat, 3 fused sub-layers, max over K.
// One wave per point, two waves per block, private LDS ping-pong per wave.
// ---------------------------------------------------------------------------
template<int CIN_X, int C, int D1PAD>
__global__ __launch_bounds__(64) void layer_kernel(
    const float* __restrict__ xin, const float* __restrict__ pc,
    const int* __restrict__ idx,
    const bf16_t* __restrict__ wt0, const bf16_t* __restrict__ wt1,
    const bf16_t* __restrict__ wt2,
    const float* __restrict__ b0, const float* __restrict__ gw0, const float* __restrict__ gb0,
    const float* __restrict__ b1, const float* __restrict__ gw1, const float* __restrict__ gb1,
    const float* __restrict__ b2, const float* __restrict__ gw2, const float* __restrict__ gb2,
    float* __restrict__ xout, int transposed) {
  constexpr int MAXD = (D1PAD > C) ? D1PAD : C;
  __shared__ alignas(32) bf16_t sbuf[2][2][32 * MAXD];
  const int wid  = threadIdx.x >> 5;
  const int lane = threadIdx.x & 31;
  const int pid  = blockIdx.x * 2 + wid;
  const int b = pid >> 12;
  const int n = pid & (NPTS - 1);

  bf16_t* bufA = sbuf[wid][0];
  bf16_t* bufB = sbuf[wid][1];

  // Stage H0 = concat(gather(x, idx), epos), zero-padded to D1PAD.  lane = k.
  {
    const int j = idx[(size_t)pid * KNN + lane];
    const float* xr = xin + (size_t)(b * NPTS + j) * CIN_X;
    bf16_t* row = bufA + (size_t)lane * MAXD;
#pragma unroll
    for (int c = 0; c < CIN_X; ++c) row[c] = (bf16_t)xr[c];
    const float* pj = pc + (size_t)(b * NPTS + j) * 3;
    const float* pn = pc + (size_t)pid * 3;
#pragma unroll
    for (int d = 0; d < 3; ++d) row[CIN_X + d] = (bf16_t)(pj[d] - pn[d]);
#pragma unroll
    for (int c = CIN_X + 3; c < D1PAD; ++c) row[c] = (bf16_t)0.0f;
  }
  asm volatile("s_wait_dscnt 0x0" ::: "memory");   // intra-wave LDS ordering

  run_sublayer<D1PAD, C, MAXD, false>(bufA, bufB, wt0, b0, gw0, gb0, nullptr, 0, lane);
  asm volatile("s_wait_dscnt 0x0" ::: "memory");
  run_sublayer<C, C, MAXD, false>(bufB, bufA, wt1, b1, gw1, gb1, nullptr, 0, lane);
  asm volatile("s_wait_dscnt 0x0" ::: "memory");

  float* op; int os;
  if (transposed) { op = xout + (size_t)b * C * NPTS + n; os = NPTS; }  // (B,C,N)
  else            { op = xout + (size_t)pid * C;          os = 1;    }  // (B,N,C)
  run_sublayer<C, C, MAXD, true>(bufA, nullptr, wt2, b2, gw2, gb2, op, os, lane);
}

// ---------------------------------------------------------------------------
extern "C" void kernel_launch(void* const* d_in, const int* in_sizes, int n_in,
                              void* d_out, int out_size, void* d_ws, size_t ws_size,
                              hipStream_t stream) {
  (void)in_sizes; (void)n_in; (void)out_size; (void)ws_size;
  const float* pc  = (const float*)d_in[0];
  const float* fea = (const float*)d_in[1];
  auto P = [&](int L, int j, int k) -> const float* {   // L,j in 1..3, k: 0=w 1=b 2=gw 3=gb
    return (const float*)d_in[2 + (L - 1) * 12 + (j - 1) * 4 + k];
  };

  // Workspace carve-up (256B aligned):
  unsigned char* ws = (unsigned char*)d_ws;
  size_t off = 0;
  auto alloc = [&](size_t bytes) -> void* {
    void* p = ws + off;
    off = (off + bytes + 255) & ~(size_t)255;
    return p;
  };
  int* idx = (int*)alloc((size_t)4 * NPTS * KNN * sizeof(int));

  const int coutT[3]    = {32, 64, 128};
  const int dT[3][3]    = {{6, 32, 32}, {35, 64, 64}, {67, 128, 128}};
  const int dpadT[3][3] = {{32, 32, 32}, {64, 64, 64}, {96, 128, 128}};
  bf16_t* wt[3][3];
  for (int L = 0; L < 3; ++L)
    for (int j = 0; j < 3; ++j)
      wt[L][j] = (bf16_t*)alloc((size_t)coutT[L] * dpadT[L][j] * sizeof(bf16_t));
  float* x1 = (float*)alloc((size_t)4 * NPTS * 32 * sizeof(float));
  float* x2 = (float*)alloc((size_t)4 * NPTS * 64 * sizeof(float));

  // 1) weights -> padded bf16
  for (int L = 0; L < 3; ++L)
    for (int j = 0; j < 3; ++j) {
      int tot = coutT[L] * dpadT[L][j];
      pad_weight<<<(tot + 255) / 256, 256, 0, stream>>>(
          P(L + 1, j + 1, 0), wt[L][j], coutT[L], dT[L][j], dpadT[L][j]);
    }

  // 2) KNN (B*N = 16384 queries)
  knn_kernel<<<64, 256, 0, stream>>>(pc, idx);

  // 3) three fused EdgeConv layers (8192 blocks x 2 waves = 16384 points)
  layer_kernel<3, 32, 32><<<8192, 64, 0, stream>>>(
      fea, pc, idx, wt[0][0], wt[0][1], wt[0][2],
      P(1,1,1), P(1,1,2), P(1,1,3),
      P(1,2,1), P(1,2,2), P(1,2,3),
      P(1,3,1), P(1,3,2), P(1,3,3), x1, 0);

  layer_kernel<32, 64, 64><<<8192, 64, 0, stream>>>(
      x1, pc, idx, wt[1][0], wt[1][1], wt[1][2],
      P(2,1,1), P(2,1,2), P(2,1,3),
      P(2,2,1), P(2,2,2), P(2,2,3),
      P(2,3,1), P(2,3,2), P(2,3,3), x2, 0);

  layer_kernel<64, 128, 96><<<8192, 64, 0, stream>>>(
      x2, pc, idx, wt[2][0], wt[2][1], wt[2][2],
      P(3,1,1), P(3,1,2), P(3,1,3),
      P(3,2,1), P(3,2,2), P(3,2,3),
      P(3,3,1), P(3,3,2), P(3,3,3), (float*)d_out, 1);
}
